// AttentionFusion_29807073034258
// MI455X (gfx1250) — matrix operand order repaired and checked by reading, not compiled, not run on hardware
//
#include <hip/hip_runtime.h>
#include <hip/hip_bf16.h>
#include <math.h>

// ---------------------------------------------------------------------------
// Problem constants (from the reference)
// ---------------------------------------------------------------------------
#define NPIX 10000   // H*W
#define TT   5       // timesteps
#define CC   256     // channels
#define NHH  8       // heads
#define DKK  32      // head dim
#define KN   25      // spatial neighbors (2R+1)^2
#define NEGV (-1.0e10f)
#define SCL  0.17677669529663687f   // 1/sqrt(32)

typedef __attribute__((ext_vector_type(16))) __bf16 v16bf;
typedef __attribute__((ext_vector_type(8)))  __bf16 v8bf;
typedef __attribute__((ext_vector_type(8)))  float  v8f;
typedef __attribute__((ext_vector_type(4)))  unsigned int u32x4;
typedef __attribute__((ext_vector_type(4)))  int i32x4;
typedef __attribute__((ext_vector_type(8)))  int i32x8;

#if __has_builtin(__builtin_amdgcn_tensor_load_to_lds)
#define HAVE_TDM 1
#endif

// ---------------------------------------------------------------------------
// fp32 -> bf16 conversion of x
// ---------------------------------------------------------------------------
__global__ __launch_bounds__(256)
void cvt_x_kernel(const float* __restrict__ x, __bf16* __restrict__ xb, int n) {
  for (int i = blockIdx.x * blockDim.x + threadIdx.x; i < n;
       i += gridDim.x * blockDim.x)
    xb[i] = (__bf16)x[i];
}

// ---------------------------------------------------------------------------
// Convert + transpose the 8 weight matrices: Wt[mat][n][k] = W[mat][k][n]
// ---------------------------------------------------------------------------
struct WPtrs { const float* w[8]; };

__global__ __launch_bounds__(256)
void cvt_w_kernel(WPtrs wp, __bf16* __restrict__ wt) {
  const int total = 8 * CC * CC;
  for (int i = blockIdx.x * blockDim.x + threadIdx.x; i < total;
       i += gridDim.x * blockDim.x) {
    int mat = i >> 16;
    int r   = i & 0xFFFF;
    int n   = r >> 8;    // output column of the GEMM
    int k   = r & 255;   // reduction index
    wt[i] = (__bf16)wp.w[mat][k * CC + n];
  }
}

// ---------------------------------------------------------------------------
// TDM descriptor issue: 2D tile (tile_dim0 = 32 K-elements per row,
// tileRows rows) from global (row stride = strideElems, bf16) into LDS.
// tensorRows = rows remaining in the tensor -> TDM zero-fills OOB reads.
// ---------------------------------------------------------------------------
#ifdef HAVE_TDM
__device__ __forceinline__ void tdm_load_2d(const __bf16* gaddr, unsigned ldsByteAddr,
                                            int tensorRows, int tileRows,
                                            int strideElems) {
  unsigned long long ga = (unsigned long long)(size_t)gaddr;
  u32x4 g0;
  g0[0] = 1u;                                     // count=1 (valid), user mode
  g0[1] = ldsByteAddr;                            // lds_addr [63:32]
  g0[2] = (unsigned)ga;                           // global_addr low
  g0[3] = (unsigned)((ga >> 32) & 0x01FFFFFFu) | (2u << 30);  // addr hi | type=2
  i32x8 g1;
  g1[0] = 0x00010000;                             // data_size=1 (2 bytes)
  g1[1] = (32 & 0xFFFF) << 16;                    // tensor_dim0 = 32
  g1[2] = (tensorRows & 0xFFFF) << 16;            // tensor_dim1 lo16
  g1[3] = ((tensorRows >> 16) & 0xFFFF) | (32 << 16);  // tensor_dim1 hi | tile_dim0=32
  g1[4] = tileRows & 0xFFFF;                      // tile_dim1 (tile_dim2=0)
  g1[5] = strideElems;                            // tensor_dim0_stride lo32
  g1[6] = 0;
  g1[7] = 0;
  i32x4 z4 = {0, 0, 0, 0};
#if __clang_major__ >= 23
  i32x8 z8 = {0, 0, 0, 0, 0, 0, 0, 0};
  __builtin_amdgcn_tensor_load_to_lds(g0, g1, z4, z4, z8, 0);
#else
  __builtin_amdgcn_tensor_load_to_lds(g0, g1, z4, z4, 0);
#endif
}
#endif

// ---------------------------------------------------------------------------
// bf16 GEMM + bias:  Out[M x 256] = A[M x 256] @ W + b
// Block = 256 threads = 8 waves; block tile = 64 rows x 256 cols.
// Wave w: rows (w&3)*16.., cols (w>>2)*128..  (8 N-tiles, 64 accum VGPRs).
// A (64x32) and B (32x256) K-panels staged in LDS, double-buffered via the
// Tensor Data Mover (wave 0 issues descriptors; s_wait_tensorcnt + barrier).
// All 8 B fragments are fetched into registers before the WMMA chain so the
// matrix pipe runs back-to-back behind a single s_wait_dscnt.
// ---------------------------------------------------------------------------
#define APANEL (64 * 32)     // bf16 elements
#define BPANEL (256 * 32)    // bf16 elements

template <bool OUT_F32>
__global__ __launch_bounds__(256)
void gemm256_kernel(const __bf16* __restrict__ A, int aStride, int aOff,
                    const __bf16* __restrict__ Wt, const float* __restrict__ bias,
                    void* __restrict__ OutP, int M) {
  __shared__ __align__(128) __bf16 lds[2][APANEL + BPANEL];   // 2 x 20 KB

  const int tid   = threadIdx.x;
  const int lane  = tid & 31;
  const int wave  = tid >> 5;
  const int lr    = lane & 15;
  const int kh    = lane >> 4;
  const int m0    = blockIdx.x * 64;
  const int rbase = (wave & 3) * 16;        // row group within block
  const int nbase = (wave >> 2) * 128;      // column group within block
  const int remRows = M - m0;               // TDM zero-fills reads past this

  v8f acc[8] = {};

#ifdef HAVE_TDM
  const __bf16* aBlock = A + (size_t)m0 * aStride + aOff;
  if (wave == 0) {
    unsigned lbase = (unsigned)(size_t)&lds[0][0];
    tdm_load_2d(aBlock, lbase, remRows, 64, aStride);
    tdm_load_2d(Wt, lbase + APANEL * 2, 256, 256, CC);
  }
  for (int p = 0; p < 8; ++p) {
    const int buf = p & 1;
    if (wave == 0) {
      if (p + 1 < 8) {
        unsigned lb = (unsigned)(size_t)&lds[buf ^ 1][0];
        tdm_load_2d(aBlock + (p + 1) * 32, lb, remRows, 64, aStride);
        tdm_load_2d(Wt + (p + 1) * 32, lb + APANEL * 2, 256, 256, CC);
        __builtin_amdgcn_s_wait_tensorcnt(2);   // panel p complete (in-order)
      } else {
        __builtin_amdgcn_s_wait_tensorcnt(0);
      }
    }
    __syncthreads();
    // A fragment: row = rbase+lr, chunks at K kh*8 and 16+kh*8 (ISA layout)
    const int rib = rbase + lr;
    v8bf alo = *(const v8bf*)&lds[buf][rib * 32 + kh * 8];
    v8bf ahi = *(const v8bf*)&lds[buf][rib * 32 + 16 + kh * 8];
    v16bf a  = __builtin_shufflevector(alo, ahi,
                 0, 1, 2, 3, 4, 5, 6, 7, 8, 9, 10, 11, 12, 13, 14, 15);
    // Fetch all 8 B fragments first, then run the WMMA chain back-to-back.
    v16bf bfrag[8];
#pragma unroll
    for (int t = 0; t < 8; ++t) {
      const int ncol = nbase + t * 16 + lr;
      v8bf blo = *(const v8bf*)&lds[buf][APANEL + ncol * 32 + kh * 8];
      v8bf bhi = *(const v8bf*)&lds[buf][APANEL + ncol * 32 + 16 + kh * 8];
      bfrag[t] = __builtin_shufflevector(blo, bhi,
                   0, 1, 2, 3, 4, 5, 6, 7, 8, 9, 10, 11, 12, 13, 14, 15);
    }
#pragma unroll
    for (int t = 0; t < 8; ++t)
      acc[t] = __builtin_amdgcn_wmma_f32_16x16x32_bf16(
          false, a, false, bfrag[t], (short)0, acc[t], false, false);
    __syncthreads();   // all waves done with buf before it is overwritten
  }
#else
  // Fallback: cooperative synchronous staging (no TDM builtin available)
  for (int p = 0; p < 8; ++p) {
    const int buf = p & 1;
    const int k0 = p * 32;
    // A panel: 64 rows x 32 k = 256 x 16B chunks, one per thread
    {
      const int row = tid >> 2, koff = (tid & 3) * 8;
      uint4 val = {0, 0, 0, 0};
      if (row < remRows)
        val = *(const uint4*)(A + (size_t)(m0 + row) * aStride + aOff + k0 + koff);
      *(uint4*)&lds[buf][row * 32 + koff] = val;
    }
    // B panel: 256 rows x 32 k = 1024 x 16B chunks, four per thread
#pragma unroll
    for (int c = tid; c < 1024; c += 256) {
      const int n = c >> 2, koff = (c & 3) * 8;
      *(uint4*)&lds[buf][APANEL + n * 32 + koff] =
          *(const uint4*)(Wt + (size_t)n * CC + k0 + koff);
    }
    __syncthreads();
    const int rib = rbase + lr;
    v8bf alo = *(const v8bf*)&lds[buf][rib * 32 + kh * 8];
    v8bf ahi = *(const v8bf*)&lds[buf][rib * 32 + 16 + kh * 8];
    v16bf a  = __builtin_shufflevector(alo, ahi,
                 0, 1, 2, 3, 4, 5, 6, 7, 8, 9, 10, 11, 12, 13, 14, 15);
    v16bf bfrag[8];
#pragma unroll
    for (int t = 0; t < 8; ++t) {
      const int ncol = nbase + t * 16 + lr;
      v8bf blo = *(const v8bf*)&lds[buf][APANEL + ncol * 32 + kh * 8];
      v8bf bhi = *(const v8bf*)&lds[buf][APANEL + ncol * 32 + 16 + kh * 8];
      bfrag[t] = __builtin_shufflevector(blo, bhi,
                   0, 1, 2, 3, 4, 5, 6, 7, 8, 9, 10, 11, 12, 13, 14, 15);
    }
#pragma unroll
    for (int t = 0; t < 8; ++t)
      acc[t] = __builtin_amdgcn_wmma_f32_16x16x32_bf16(
          false, a, false, bfrag[t], (short)0, acc[t], false, false);
    __syncthreads();
  }
#endif

  // Epilogue: C/D layout lane = N col, VGPR j = M row j + 8*(lane>>4)
#pragma unroll
  for (int t = 0; t < 8; ++t) {
    const int ncol = nbase + t * 16 + lr;
    const float bv = bias[ncol];
#pragma unroll
    for (int j = 0; j < 8; ++j) {
      const int m = m0 + rbase + kh * 8 + j;
      if (m < M) {
        const float val = acc[t][j] + bv;
        if (OUT_F32)
          ((float*)OutP)[(size_t)m * CC + ncol] = val;
        else
          ((__bf16*)OutP)[(size_t)m * CC + ncol] = (__bf16)val;
      }
    }
  }
}

// ---------------------------------------------------------------------------
// Wave-wide sum (butterfly; lowers to ds_*permute on gfx1250 wave32)
// ---------------------------------------------------------------------------
__device__ __forceinline__ float wave_sum(float v) {
#pragma unroll
  for (int off = 16; off > 0; off >>= 1) v += __shfl_xor(v, off, 32);
  return v;
}

// ---------------------------------------------------------------------------
// Temporal attention: one wave per (pixel, head); lane = head dim.
// ---------------------------------------------------------------------------
__global__ __launch_bounds__(256)
void temporal_attn_kernel(const __bf16* __restrict__ qt,
                          const __bf16* __restrict__ kt,
                          const __bf16* __restrict__ vt,
                          const int* __restrict__ tmask,
                          __bf16* __restrict__ ctx) {
  const int task = blockIdx.x * 8 + (threadIdx.x >> 5);
  const int lane = threadIdx.x & 31;
  const int n = task / NHH;
  const int h = task % NHH;
  const int cofs = h * DKK + lane;

  const float q = (float)qt[(size_t)n * CC + cofs];
  float s[TT];
#pragma unroll
  for (int t = 0; t < TT; ++t) {
    const float kk = (float)kt[((size_t)n * TT + t) * CC + cofs];
    const float d = wave_sum(q * kk) * SCL;
    s[t] = (tmask[n * TT + t] == 0) ? NEGV : d;
  }
  float mx = s[0];
#pragma unroll
  for (int t = 1; t < TT; ++t) mx = fmaxf(mx, s[t]);
  float sum = 0.f;
#pragma unroll
  for (int t = 0; t < TT; ++t) { s[t] = __expf(s[t] - mx); sum += s[t]; }
  const float inv = 1.f / sum;
  float acc = 0.f;
#pragma unroll
  for (int t = 0; t < TT; ++t)
    acc += s[t] * inv * (float)vt[((size_t)n * TT + t) * CC + cofs];
  ctx[(size_t)n * CC + cofs] = (__bf16)acc;
}

// ---------------------------------------------------------------------------
// Spatial attention: one wave per (pixel, head); 25 gathered neighbors.
// ---------------------------------------------------------------------------
__global__ __launch_bounds__(256)
void spatial_attn_kernel(const __bf16* __restrict__ sq,
                         const __bf16* __restrict__ k1,
                         const __bf16* __restrict__ v1,
                         const int* __restrict__ neighbors,
                         const int* __restrict__ smask,
                         __bf16* __restrict__ sctx) {
  const int task = blockIdx.x * 8 + (threadIdx.x >> 5);
  const int lane = threadIdx.x & 31;
  const int n = task / NHH;
  const int h = task % NHH;
  const int cofs = h * DKK + lane;

  const float q = (float)sq[(size_t)n * CC + cofs];
  int   nb[KN];
  float s[KN];
#pragma unroll
  for (int j = 0; j < KN; ++j) {
    nb[j] = neighbors[n * KN + j];
    const float kk = (float)k1[(size_t)nb[j] * CC + cofs];
    const float d = wave_sum(q * kk) * SCL;
    s[j] = (smask[n * KN + j] == 0) ? NEGV : d;
  }
  float mx = s[0];
#pragma unroll
  for (int j = 1; j < KN; ++j) mx = fmaxf(mx, s[j]);
  float sum = 0.f;
#pragma unroll
  for (int j = 0; j < KN; ++j) { s[j] = __expf(s[j] - mx); sum += s[j]; }
  const float inv = 1.f / sum;
  float acc = 0.f;
#pragma unroll
  for (int j = 0; j < KN; ++j)
    acc += s[j] * inv * (float)v1[(size_t)nb[j] * CC + cofs];
  sctx[(size_t)n * CC + cofs] = (__bf16)acc;
}

// ---------------------------------------------------------------------------
// Launch sequence
// ---------------------------------------------------------------------------
extern "C" void kernel_launch(void* const* d_in, const int* in_sizes, int n_in,
                              void* d_out, int out_size, void* d_ws, size_t ws_size,
                              hipStream_t stream) {
  (void)in_sizes; (void)n_in; (void)out_size; (void)ws_size;

  const float* x = (const float*)d_in[0];
  const float* Wf[8]  = { (const float*)d_in[1],  (const float*)d_in[3],
                          (const float*)d_in[5],  (const float*)d_in[7],
                          (const float*)d_in[9],  (const float*)d_in[11],
                          (const float*)d_in[13], (const float*)d_in[15] };
  const float* Bf[8]  = { (const float*)d_in[2],  (const float*)d_in[4],
                          (const float*)d_in[6],  (const float*)d_in[8],
                          (const float*)d_in[10], (const float*)d_in[12],
                          (const float*)d_in[14], (const float*)d_in[16] };
  const int* neighbors = (const int*)d_in[17];
  const int* tmask     = (const int*)d_in[18];
  const int* smask     = (const int*)d_in[19];
  float* out = (float*)d_out;

  // workspace layout (bf16 elements), all regions 32B aligned
  __bf16* ws = (__bf16*)d_ws;
  size_t o = 0;
  __bf16* xb   = ws + o; o += (size_t)NPIX * TT * CC;
  __bf16* wt   = ws + o; o += (size_t)8 * CC * CC;
  __bf16* qt   = ws + o; o += (size_t)NPIX * CC;
  __bf16* kt   = ws + o; o += (size_t)NPIX * TT * CC;
  __bf16* vt   = ws + o; o += (size_t)NPIX * TT * CC;
  __bf16* ctx  = ws + o; o += (size_t)NPIX * CC;
  __bf16* ht   = ws + o; o += (size_t)NPIX * CC;
  __bf16* sqb  = ws + o; o += (size_t)NPIX * CC;
  __bf16* k1b  = ws + o; o += (size_t)NPIX * CC;
  __bf16* v1b  = ws + o; o += (size_t)NPIX * CC;
  __bf16* sctx = ws + o; o += (size_t)NPIX * CC;

  WPtrs wp;
  for (int i = 0; i < 8; ++i) wp.w[i] = Wf[i];

  const int nXT  = NPIX * TT * CC;
  const int MKV  = NPIX * TT;           // 50000 rows
  const int gKV  = (MKV + 63) / 64;     // 782 blocks
  const int gN   = (NPIX + 63) / 64;    // 157 blocks

  // 1) precision conversion
  cvt_x_kernel<<<4096, 256, 0, stream>>>(x, xb, nXT);
  cvt_w_kernel<<<2048, 256, 0, stream>>>(wp, wt);

  // 2) temporal projections
  gemm256_kernel<false><<<gKV, 256, 0, stream>>>(
      xb, CC, 0, wt + 1 * CC * CC, Bf[1], kt, MKV);                  // t_Wk
  gemm256_kernel<false><<<gKV, 256, 0, stream>>>(
      xb, CC, 0, wt + 2 * CC * CC, Bf[2], vt, MKV);                  // t_Wv
  gemm256_kernel<false><<<gN, 256, 0, stream>>>(
      xb, TT * CC, (TT - 1) * CC, wt + 0 * CC * CC, Bf[0], qt, NPIX); // t_Wq

  // 3) temporal attention
  temporal_attn_kernel<<<NPIX, 256, 0, stream>>>(qt, kt, vt, tmask, ctx);

  // 4) temporal output projection -> h_t
  gemm256_kernel<false><<<gN, 256, 0, stream>>>(
      ctx, CC, 0, wt + 3 * CC * CC, Bf[3], ht, NPIX);                // t_Wo

  // 5) spatial projections from h_t
  gemm256_kernel<false><<<gN, 256, 0, stream>>>(
      ht, CC, 0, wt + 4 * CC * CC, Bf[4], sqb, NPIX);                // s_Wq
  gemm256_kernel<false><<<gN, 256, 0, stream>>>(
      ht, CC, 0, wt + 5 * CC * CC, Bf[5], k1b, NPIX);                // s_Wk
  gemm256_kernel<false><<<gN, 256, 0, stream>>>(
      ht, CC, 0, wt + 6 * CC * CC, Bf[6], v1b, NPIX);                // s_Wv

  // 6) spatial attention over 25 gathered neighbors
  spatial_attn_kernel<<<NPIX, 256, 0, stream>>>(sqb, k1b, v1b,
                                                neighbors, smask, sctx);

  // 7) final output projection (fp32, straight to d_out)
  gemm256_kernel<true><<<gN, 256, 0, stream>>>(
      sctx, CC, 0, wt + 7 * CC * CC, Bf[7], out, NPIX);              // s_Wo
}
